// GFM_20675972563730
// MI455X (gfx1250) — compile-verified, matrix operand order checked by reading, not run on hardware
//
#include <hip/hip_runtime.h>
#include <hip/hip_bf16.h>

// ---------------------------------------------------------------------------
// GFM point-cloud transformer for gfx1250 (MI455X), wave32 + WMMA bf16.
// ---------------------------------------------------------------------------

typedef __attribute__((ext_vector_type(16))) __bf16 bf16x16;
typedef __attribute__((ext_vector_type(8)))  float  f32x8;

#define BB 16
#define NN 1024
#define KK 20
#define HH 4
#define TD 64
#define BN (BB * NN)   // 16384 rows

__device__ __forceinline__ f32x8 wmma_bf16(bf16x16 a, bf16x16 b, f32x8 c) {
  // D = A(16x32) * B(32x16) + C, f32 accumulate
  return __builtin_amdgcn_wmma_f32_16x16x32_bf16(false, a, false, b,
                                                 (short)0, c, false, false);
}

// ---------------------------------------------------------------------------
// kNN: per (b,n) keep 20 smallest squared distances (== top-k of exp(-d2/2)).
// Points tiled through LDS. Tie-break: earlier index wins (matches top_k).
// ---------------------------------------------------------------------------
__global__ void knn_kernel(const float* __restrict__ x, int* __restrict__ idx) {
  __shared__ float spx[256], spy[256], spz[256];
  const int b = blockIdx.y;
  const int n = blockIdx.x * 256 + threadIdx.x;
  const float* xb = x + (size_t)b * 3 * NN;
  const float px = xb[n], py = xb[NN + n], pz = xb[2 * NN + n];

  float bd[KK]; int bi[KK];
#pragma unroll
  for (int k = 0; k < KK; ++k) { bd[k] = 1e30f; bi[k] = 0; }

  for (int t = 0; t < NN; t += 256) {
    __syncthreads();
    spx[threadIdx.x] = xb[t + threadIdx.x];
    spy[threadIdx.x] = xb[NN + t + threadIdx.x];
    spz[threadIdx.x] = xb[2 * NN + t + threadIdx.x];
    __syncthreads();
    for (int m = 0; m < 256; ++m) {
      const float dx = px - spx[m], dy = py - spy[m], dz = pz - spz[m];
      const float d2 = dx * dx + dy * dy + dz * dz;
      if (d2 < bd[KK - 1]) {
        int j = KK - 1;
        while (j > 0 && bd[j - 1] > d2) { bd[j] = bd[j - 1]; bi[j] = bi[j - 1]; --j; }
        bd[j] = d2; bi[j] = t + m;
      }
    }
  }
  int* op = idx + ((size_t)b * NN + n) * KK;
#pragma unroll
  for (int k = 0; k < KK; ++k) op[k] = bi[k];
}

// ---------------------------------------------------------------------------
// 3-point orthonormal DCT-II along channel dim: [B,C,N] -> [B*N,3]
// ---------------------------------------------------------------------------
__global__ void dct_kernel(const float* __restrict__ x, float* __restrict__ xd) {
  const int i = blockIdx.x * blockDim.x + threadIdx.x;  // b*N + n
  if (i >= BN) return;
  const int b = i >> 10, n = i & (NN - 1);
  const float* xb = x + (size_t)b * 3 * NN;
  const float p0 = xb[n], p1 = xb[NN + n], p2 = xb[2 * NN + n];
  xd[(size_t)i * 3 + 0] = 0.5773502691896258f * (p0 + p1 + p2);
  xd[(size_t)i * 3 + 1] = 0.7071067811865476f * (p0 - p2);
  xd[(size_t)i * 3 + 2] = 0.8164965809277260f * (0.5f * p0 - p1 + 0.5f * p2);
}

// ---------------------------------------------------------------------------
// gather: out[b,n,k*D+d] = in[(b*N + idx[b,n,k]) * ld_in + col_off + d]
// ---------------------------------------------------------------------------
__global__ void gather_kernel(const float* __restrict__ in, const int* __restrict__ idx,
                              float* __restrict__ out, int D, int ld_in, int col_off) {
  const long total = (long)BN * KK * D;
  for (long t = blockIdx.x * (long)blockDim.x + threadIdx.x; t < total;
       t += (long)gridDim.x * blockDim.x) {
    const int  d  = (int)(t % D);
    const long nk = t / D;
    const int  k  = (int)(nk % KK);
    const long bn = nk / KK;
    const int  b  = (int)(bn >> 10);
    const int  src = idx[bn * KK + k];
    out[t] = in[((size_t)b * NN + src) * ld_in + col_off + d];
  }
}

// ---------------------------------------------------------------------------
// GEMM: Y[M,64] = act(X[M,Kd] @ W[Kd,64] + bias).
// ACT: 0=none, 1=tanh-GELU.  OBF: 0 -> f32 output, 1 -> bf16 output.
// Block = 128 threads (4 waves); wave w owns output cols [16w,16w+16);
// each block computes a 16x64 strip.  K-loop step 32 with zero-pad tail.
// ---------------------------------------------------------------------------
template<int ACT, int OBF>
__global__ void gemm64(const float* __restrict__ X, const float* __restrict__ W,
                       const float* __restrict__ bias, void* __restrict__ Yv,
                       int Kd) {
  const int wave = threadIdx.x >> 5;
  const int lane = threadIdx.x & 31;
  const int half = lane >> 4;
  const int l16  = lane & 15;
  const int row0 = blockIdx.x * 16;
  const int col  = wave * 16 + l16;

  f32x8 acc = {};
  for (int k0 = 0; k0 < Kd; k0 += 32) {
    if (k0 + 32 < Kd)   // prefetch next A tile (global_prefetch_b8)
      __builtin_prefetch(&X[(size_t)(row0 + l16) * Kd + k0 + 32], 0, 1);
    bf16x16 a, b;
#pragma unroll
    for (int e = 0; e < 8; ++e) {
      const int ka = k0 + half * 8 + e;         // K for elements 0..7
      const int kh = k0 + 16 + half * 8 + e;    // K for elements 8..15
      a[e]     = (ka < Kd) ? (__bf16)X[(size_t)(row0 + l16) * Kd + ka] : (__bf16)0.f;
      a[e + 8] = (kh < Kd) ? (__bf16)X[(size_t)(row0 + l16) * Kd + kh] : (__bf16)0.f;
      b[e]     = (ka < Kd) ? (__bf16)W[(size_t)ka * TD + col] : (__bf16)0.f;
      b[e + 8] = (kh < Kd) ? (__bf16)W[(size_t)kh * TD + col] : (__bf16)0.f;
    }
    acc = wmma_bf16(a, b, acc);
  }

  const float bc = bias[col];
#pragma unroll
  for (int r = 0; r < 8; ++r) {
    float v = acc[r] + bc;
    if (ACT == 1) {  // jax.nn.gelu (approximate=True)
      const float u = 0.7978845608028654f * (v + 0.044715f * v * v * v);
      v = 0.5f * v * (1.0f + tanhf(u));
    }
    const size_t o = (size_t)(row0 + r + 8 * half) * TD + col;
    if (OBF) ((__bf16*)Yv)[o] = (__bf16)v;
    else     ((float*)Yv)[o]  = v;
  }
}

// ---------------------------------------------------------------------------
// Attention, one wave per (b, h, 16-query tile).  dh=16 padded to K=32.
// Computes S' = K_tile * Q_tile^T so each lane's 8 C-tile slots hold 8 kv
// scores of ONE query: softmax reductions are in-register (+1 shfl), and the
// S' layout IS the A-fragment layout of P for O += P*V (no LDS transpose).
// Exact two-pass softmax per query tile (pass1 max, pass2 exp-sum + PV).
// ---------------------------------------------------------------------------
__global__ void attn_kernel(const __bf16* __restrict__ Q, const __bf16* __restrict__ Km,
                            const __bf16* __restrict__ V, float* __restrict__ O) {
  __shared__ float lds_rl[16];       // per-query 1/l broadcast at the end
  const int lane = threadIdx.x & 31;
  const int half = lane >> 4;
  const int l16  = lane & 15;
  const int bh   = blockIdx.y;
  const int b    = bh >> 2;          // H = 4
  const int h    = bh & 3;
  const int q0   = b * NN + blockIdx.x * 16;
  const int co   = h * 16;           // head column offset, dh = 16

  // B-frag of Q^T (K-dim = dh padded to 32, N = query): loaded once.
  bf16x16 qb;
#pragma unroll
  for (int e = 0; e < 8; ++e) {
    qb[e]     = Q[(size_t)(q0 + l16) * TD + co + half * 8 + e];
    qb[e + 8] = (__bf16)0.f;
  }
  const f32x8 zero = {};

  // ---- pass 1: per-query max over all kv ----
  float m = -1e30f;
  for (int j = 0; j < NN / 16; ++j) {
    const int k0 = b * NN + j * 16;
    bf16x16 ka;                      // A-frag of K tile (M = kv, K = dh pad)
#pragma unroll
    for (int e = 0; e < 8; ++e) {
      ka[e]     = Km[(size_t)(k0 + l16) * TD + co + half * 8 + e];
      ka[e + 8] = (__bf16)0.f;
    }
    f32x8 s = wmma_bf16(ka, qb, zero);   // slot r = S[kv=r+8*half][q=l16]
#pragma unroll
    for (int r = 0; r < 8; ++r) m = fmaxf(m, s[r] * 0.25f);
  }
  m = fmaxf(m, __shfl_xor(m, 16, 32));   // combine kv halves (same query)

  // ---- pass 2: exp-sum and O += P * V ----
  float l = 0.f;
  f32x8 accO = {};
  for (int j = 0; j < NN / 16; ++j) {
    const int k0 = b * NN + j * 16;
    bf16x16 ka, vb;
#pragma unroll
    for (int e = 0; e < 8; ++e) {
      ka[e]     = Km[(size_t)(k0 + l16) * TD + co + half * 8 + e];
      ka[e + 8] = (__bf16)0.f;
      vb[e]     = V[(size_t)(k0 + half * 8 + e) * TD + co + l16];   // B[K=kv][n=dh]
      vb[e + 8] = (__bf16)0.f;
    }
    f32x8 s = wmma_bf16(ka, qb, zero);
    bf16x16 pa;                      // A-frag of P: identical layout to s!
#pragma unroll
    for (int r = 0; r < 8; ++r) {
      const float p = __expf(s[r] * 0.25f - m);
      l += p;
      pa[r] = (__bf16)p;
    }
#pragma unroll
    for (int r = 0; r < 8; ++r) pa[r + 8] = (__bf16)0.f;   // kv 16..31 pad
    accO = wmma_bf16(pa, vb, accO);  // D rows = q, cols = dh
  }
  l += __shfl_xor(l, 16, 32);

  // broadcast per-query 1/l via LDS (lane q on half 0 owns query q)
  if (lane < 16) lds_rl[lane] = 1.f / l;
  __syncthreads();
#pragma unroll
  for (int r = 0; r < 8; ++r)
    O[(size_t)(q0 + r + 8 * half) * TD + co + l16] = accO[r] * lds_rl[r + 8 * half];
}

// ---------------------------------------------------------------------------
// out[row, col_off..col_off+64) = LayerNorm64(A + C) * g + beta.
// One wave per row (2 elems/lane), shuffle reductions.
// ---------------------------------------------------------------------------
__global__ void resln64(const float* __restrict__ A, const float* __restrict__ C,
                        const float* __restrict__ g, const float* __restrict__ be,
                        float* __restrict__ out, int ld_out, int col_off) {
  const int row  = blockIdx.x * (blockDim.x >> 5) + (threadIdx.x >> 5);
  const int lane = threadIdx.x & 31;
  float x0 = A[(size_t)row * TD + lane]      + C[(size_t)row * TD + lane];
  float x1 = A[(size_t)row * TD + 32 + lane] + C[(size_t)row * TD + 32 + lane];
  float s = x0 + x1;
#pragma unroll
  for (int o = 1; o < 32; o <<= 1) s += __shfl_xor(s, o, 32);
  const float mu = s * (1.f / 64.f);
  const float v0 = x0 - mu, v1 = x1 - mu;
  float vs = v0 * v0 + v1 * v1;
#pragma unroll
  for (int o = 1; o < 32; o <<= 1) vs += __shfl_xor(vs, o, 32);
  const float inv = rsqrtf(vs * (1.f / 64.f) + 1e-5f);
  out[(size_t)row * ld_out + col_off + lane]      = v0 * inv * g[lane]      + be[lane];
  out[(size_t)row * ld_out + col_off + 32 + lane] = v1 * inv * g[32 + lane] + be[32 + lane];
}

// ---------------------------------------------------------------------------
// leaky_relu(0.01) + LayerNorm256 + transpose to [B, 256, N].
// ---------------------------------------------------------------------------
__global__ void final_ln(const float* __restrict__ comb, const float* __restrict__ g,
                         const float* __restrict__ be, float* __restrict__ out) {
  const int row  = blockIdx.x * (blockDim.x >> 5) + (threadIdx.x >> 5);  // b*N + n
  const int lane = threadIdx.x & 31;
  float v[8];
  float s = 0.f;
#pragma unroll
  for (int e = 0; e < 8; ++e) {
    float x = comb[(size_t)row * 256 + e * 32 + lane];
    x = (x > 0.f) ? x : 0.01f * x;
    v[e] = x; s += x;
  }
#pragma unroll
  for (int o = 1; o < 32; o <<= 1) s += __shfl_xor(s, o, 32);
  const float mu = s * (1.f / 256.f);
  float vs = 0.f;
#pragma unroll
  for (int e = 0; e < 8; ++e) { const float d = v[e] - mu; vs += d * d; }
#pragma unroll
  for (int o = 1; o < 32; o <<= 1) vs += __shfl_xor(vs, o, 32);
  const float inv = rsqrtf(vs * (1.f / 256.f) + 1e-5f);
  const int b = row >> 10, n = row & (NN - 1);
#pragma unroll
  for (int e = 0; e < 8; ++e) {
    const int d = e * 32 + lane;
    out[(size_t)b * 256 * NN + (size_t)d * NN + n] = (v[e] - mu) * inv * g[d] + be[d];
  }
}

// ---------------------------------------------------------------------------
extern "C" void kernel_launch(void* const* d_in, const int* in_sizes, int n_in,
                              void* d_out, int out_size, void* d_ws, size_t ws_size,
                              hipStream_t stream) {
  const float* x     = (const float*)d_in[0];
  const float* W_in0 = (const float*)d_in[1];
  const float* b_in0 = (const float*)d_in[2];
  const float* W_in  = (const float*)d_in[3];
  const float* b_in  = (const float*)d_in[4];
  const float* Wq = (const float*)d_in[5];  const float* bq = (const float*)d_in[6];
  const float* Wk = (const float*)d_in[7];  const float* bk = (const float*)d_in[8];
  const float* Wv = (const float*)d_in[9];  const float* bv = (const float*)d_in[10];
  const float* Wo = (const float*)d_in[11]; const float* bo = (const float*)d_in[12];
  const float* ln1g = (const float*)d_in[13]; const float* ln1b = (const float*)d_in[14];
  const float* W1 = (const float*)d_in[15]; const float* b1 = (const float*)d_in[16];
  const float* W2 = (const float*)d_in[17]; const float* b2 = (const float*)d_in[18];
  const float* ln2g = (const float*)d_in[19]; const float* ln2b = (const float*)d_in[20];
  const float* ng = (const float*)d_in[21]; const float* nb = (const float*)d_in[22];

  float* ws = (float*)d_ws;
  float* xd   = ws;                ws += (size_t)BN * 3;
  int*   idx  = (int*)ws;          ws += (size_t)BN * KK;
  float* g0   = ws;                ws += (size_t)BN * 60;
  float* gbuf = ws;                ws += (size_t)BN * 1280;
  float* Hbuf = ws;                ws += (size_t)BN * TD;
  float* H2   = ws;                ws += (size_t)BN * TD;
  __bf16* Qb  = (__bf16*)ws;       ws += (size_t)BN * TD;   // bf16, half used
  __bf16* Kb  = (__bf16*)ws;       ws += (size_t)BN * TD;
  __bf16* Vb  = (__bf16*)ws;       ws += (size_t)BN * TD;
  float* Ob   = ws;                ws += (size_t)BN * TD;
  float* T1   = ws;                ws += (size_t)BN * TD;
  float* T2   = ws;                ws += (size_t)BN * TD;
  float* comb = ws;                ws += (size_t)BN * 256;

  knn_kernel<<<dim3(NN / 256, BB), 256, 0, stream>>>(x, idx);
  dct_kernel<<<BN / 256, 256, 0, stream>>>(x, xd);
  gather_kernel<<<2048, 256, 0, stream>>>(xd, idx, g0, 3, 3, 0);   // [BN,60]

  for (int i = 0; i < 4; ++i) {
    if (i == 0) gather_kernel<<<4096, 256, 0, stream>>>(g0,   idx, gbuf, 60, 60,  0);
    else        gather_kernel<<<4096, 256, 0, stream>>>(comb, idx, gbuf, TD, 256, 64 * (i - 1));

    const int    Kd  = (i == 0) ? 1200 : 1280;
    const float* Win = (i == 0) ? W_in0 : W_in + (size_t)(i - 1) * 1280 * TD;
    const float* bin = (i == 0) ? b_in0 : b_in + (size_t)(i - 1) * TD;

    gemm64<0, 0><<<BN / 16, 128, 0, stream>>>(gbuf, Win, bin, Hbuf, Kd);
    gemm64<0, 1><<<BN / 16, 128, 0, stream>>>(Hbuf, Wq + (size_t)i * TD * TD, bq + i * TD, Qb, TD);
    gemm64<0, 1><<<BN / 16, 128, 0, stream>>>(Hbuf, Wk + (size_t)i * TD * TD, bk + i * TD, Kb, TD);
    gemm64<0, 1><<<BN / 16, 128, 0, stream>>>(Hbuf, Wv + (size_t)i * TD * TD, bv + i * TD, Vb, TD);

    attn_kernel<<<dim3(NN / 16, BB * HH), 32, 0, stream>>>(Qb, Kb, Vb, Ob);

    gemm64<0, 0><<<BN / 16, 128, 0, stream>>>(Ob, Wo + (size_t)i * TD * TD, bo + i * TD, T1, TD);
    resln64<<<BN / 8, 256, 0, stream>>>(Hbuf, T1, ln1g + i * TD, ln1b + i * TD, H2, TD, 0);

    gemm64<1, 0><<<BN / 16, 128, 0, stream>>>(H2, W1 + (size_t)i * TD * TD, b1 + i * TD, T1, TD);
    gemm64<0, 0><<<BN / 16, 128, 0, stream>>>(T1, W2 + (size_t)i * TD * TD, b2 + i * TD, T2, TD);
    resln64<<<BN / 8, 256, 0, stream>>>(H2, T2, ln2g + i * TD, ln2b + i * TD, comb, 256, 64 * i);
  }

  final_ln<<<BN / 8, 256, 0, stream>>>(comb, ng, nb, (float*)d_out);
  (void)in_sizes; (void)n_in; (void)out_size; (void)ws_size;
}
